// BitLinear_2525440770142
// MI455X (gfx1250) — matrix operand order repaired and checked by reading, not compiled, not run on hardware
//
#include <hip/hip_runtime.h>
#include <hip/hip_bf16.h>
#include <stdint.h>

#define B_  4
#define S_  2048
#define K_  2048
#define N_  8192
#define M_  (B_ * S_)        // 8192

#define BT   128             // block tile (M and N)
#define KT   32              // K step per WMMA
#define LDSS 40              // padded LDS row stride in bf16 elements (80B)
#define TILE_ELEMS (BT * LDSS)   // one padded tile in LDS (bf16 elems)

typedef __attribute__((ext_vector_type(16))) __bf16 v16bf;
typedef __attribute__((ext_vector_type(8)))  __bf16 v8bf;
typedef __attribute__((ext_vector_type(8)))  float  v8f;
typedef __attribute__((ext_vector_type(4)))  unsigned int v4u;
typedef __attribute__((ext_vector_type(8)))  unsigned int v8u;

__device__ __forceinline__ unsigned short f2bf(float f) {
  union { float f; unsigned u; } cv; cv.f = f;
  unsigned u = cv.u;
  u += 0x7FFFu + ((u >> 16) & 1u);   // round-to-nearest-even
  return (unsigned short)(u >> 16);
}

// ---------------------------------------------------------------------------
// TDM: DMA one [128 rows x 32 bf16] tile from a row-major [*, K_] matrix in
// global memory into LDS at byte offset lds_off, inserting 16B of padding
// after every 64B row => LDS row stride 80B (= LDSS bf16 elements).
// 2D tensor: descriptor groups 2/3 are NULL (2-operand instruction form).
// ---------------------------------------------------------------------------
__device__ __forceinline__ void tdm_load_tile(unsigned lds_off,
                                              const unsigned short* gptr) {
  unsigned long long ga = (unsigned long long)(uintptr_t)gptr;

  v4u g0;
  g0[0] = 1u;                                   // count=1, user mode
  g0[1] = lds_off;                              // lds_addr (bytes)
  g0[2] = (unsigned)ga;                         // global_addr[31:0]
  g0[3] = (unsigned)((ga >> 32) & 0x01FFFFFFu)  // global_addr[56:32]
        | 0x80000000u;                          // type=2 ("image")

  v8u g1;
  g1[0] = (1u << 16)                            // data_size = 2 bytes
        | (1u << 20)                            // pad_enable
        | (3u << 22)                            // pad_interval: 16 DW = 64B
        | (3u << 25);                           // pad_amount:    4 DW = 16B
  g1[1] = ((unsigned)K_ & 0xFFFFu) << 16;       // tensor_dim0[15:0]
  g1[2] = (((unsigned)K_ >> 16) & 0xFFFFu)      // tensor_dim0[31:16]
        | ((8192u & 0xFFFFu) << 16);            // tensor_dim1[15:0]
  g1[3] = (8192u >> 16)                         // tensor_dim1[31:16]
        | ((unsigned)KT << 16);                 // tile_dim0 = 32 elems
  g1[4] = (unsigned)BT;                         // tile_dim1 = 128, tile_dim2=0
  g1[5] = (unsigned)K_;                         // tensor_dim0_stride[31:0]
  g1[6] = 0u;                                   // stride0 hi, stride1 lo
  g1[7] = 0u;                                   // stride1 hi

  asm volatile("tensor_load_to_lds %0, %1" :: "s"(g0), "s"(g1) : "memory");
}

// ---------------------------------------------------------------------------
// Kernel 1: per-row scale + ternary quantization of weight -> bf16 {-1,0,1}
// ---------------------------------------------------------------------------
__global__ __launch_bounds__(256)
void bitlinear_quant_w(const float* __restrict__ w,
                       unsigned short* __restrict__ wq,
                       float* __restrict__ scale) {
  __shared__ float red[256];
  __shared__ float sc_sh;
  const int row = blockIdx.x;
  const int t   = threadIdx.x;
  const float* wr = w + (size_t)row * K_;

  float4 a = ((const float4*)wr)[t * 2 + 0];
  float4 b = ((const float4*)wr)[t * 2 + 1];
  float s = fabsf(a.x) + fabsf(a.y) + fabsf(a.z) + fabsf(a.w)
          + fabsf(b.x) + fabsf(b.y) + fabsf(b.z) + fabsf(b.w);
  red[t] = s;
  __syncthreads();
  for (int off = 128; off > 0; off >>= 1) {
    if (t < off) red[t] += red[t + off];
    __syncthreads();
  }
  if (t == 0) {
    float sc = fmaxf(red[0] * (1.0f / K_), 1e-5f);
    sc_sh = sc;
    scale[row] = sc;
  }
  __syncthreads();
  const float inv = 1.0f / sc_sh;

  float v[8] = {a.x, a.y, a.z, a.w, b.x, b.y, b.z, b.w};
  unsigned short q[8];
#pragma unroll
  for (int i = 0; i < 8; ++i) {
    float qq = rintf(v[i] * inv);
    qq = fminf(1.0f, fmaxf(-1.0f, qq));
    q[i] = f2bf(qq);                 // exact: {-1, 0, 1}
  }
  *(uint4*)(wq + (size_t)row * K_ + t * 8) = *(const uint4*)q;
}

// ---------------------------------------------------------------------------
// Kernel 2: x fp32 -> bf16 (RNE). 8 elements per thread.
// ---------------------------------------------------------------------------
__global__ __launch_bounds__(256)
void bitlinear_cvt_x(const float* __restrict__ x,
                     unsigned short* __restrict__ xq) {
  size_t i = ((size_t)blockIdx.x * blockDim.x + threadIdx.x) * 8;
  float4 a = ((const float4*)(x + i))[0];
  float4 b = ((const float4*)(x + i))[1];
  unsigned short q[8] = { f2bf(a.x), f2bf(a.y), f2bf(a.z), f2bf(a.w),
                          f2bf(b.x), f2bf(b.y), f2bf(b.z), f2bf(b.w) };
  *(uint4*)(xq + i) = *(const uint4*)q;
}

// ---------------------------------------------------------------------------
// Kernel 3: C[M,N] = Xbf16[M,K] * Wq[N,K]^T, then *= scale[n].
// 128x128 block tile, 8 waves (2Mx4N), each wave 64x32 => 4x2 wmma accums.
// Double-buffered LDS staged by the Tensor Data Mover (wave 0 issues).
// ---------------------------------------------------------------------------
__global__ __launch_bounds__(256)
void bitlinear_gemm(const unsigned short* __restrict__ xq,
                    const unsigned short* __restrict__ wq,
                    const float* __restrict__ scale,
                    float* __restrict__ out) {
  __shared__ __align__(16) unsigned short lds_a[2 * TILE_ELEMS];
  __shared__ __align__(16) unsigned short lds_b[2 * TILE_ELEMS];

  const int t    = threadIdx.x;
  const int lane = t & 31;
  const int wave = t >> 5;          // 0..7
  const int wm   = wave >> 2;       // 0..1 : 64-row slab
  const int wn   = wave & 3;        // 0..3 : 32-col slab
  const int m0   = blockIdx.y * BT;
  const int n0   = blockIdx.x * BT;

  const int halfsel = lane >> 4;    // 0: lanes 0-15, 1: lanes 16-31
  const int l16     = lane & 15;

  // uniform "this is wave 0" predicate -> true scalar branch
  const bool issuer = (__builtin_amdgcn_readfirstlane(t >> 5) == 0);

  // LDS byte addresses for the TDM descriptors (flat addr low 32 bits)
  const unsigned a_off0 = (unsigned)(uintptr_t)(&lds_a[0]);
  const unsigned b_off0 = (unsigned)(uintptr_t)(&lds_b[0]);
  const unsigned tile_bytes = TILE_ELEMS * 2;

  const unsigned short* ga = xq + (size_t)m0 * K_;   // A rows m0..m0+127
  const unsigned short* gb = wq + (size_t)n0 * K_;   // B rows n0..n0+127

  v8f acc[4][2] = {};

  // Prologue: DMA first K tile into buffer 0.
  if (issuer) {
    tdm_load_tile(a_off0, ga);
    tdm_load_tile(b_off0, gb);
    __builtin_amdgcn_s_wait_tensorcnt(0);
  }
  __syncthreads();

  for (int kk = 0; kk < K_; kk += KT) {
    const int buf = (kk >> 5) & 1;

    // Kick off next tile's DMA into the other buffer (its previous readers
    // all passed the barrier that ended the last iteration).
    if (issuer && (kk + KT < K_)) {
      tdm_load_tile(a_off0 + (buf ^ 1) * tile_bytes, ga + (kk + KT));
      tdm_load_tile(b_off0 + (buf ^ 1) * tile_bytes, gb + (kk + KT));
    }

    const unsigned short* la = lds_a + buf * TILE_ELEMS;
    const unsigned short* lb = lds_b + buf * TILE_ELEMS;

    // A fragments: row m = wm*64 + mt*16 + l16; lanes 0-15 hold K 0-7/16-23,
    // lanes 16-31 hold K 8-15/24-31.
    union { v16bf v; v8bf h[2]; } afrag[4];
#pragma unroll
    for (int mt = 0; mt < 4; ++mt) {
      const unsigned short* p =
          la + (wm * 64 + mt * 16 + l16) * LDSS + halfsel * 8;
      afrag[mt].h[0] = *(const v8bf*)(p);
      afrag[mt].h[1] = *(const v8bf*)(p + 16);
    }
    // B fragments: column n = wn*32 + nt*16 + l16; lanes 0-15 hold K 0-7/16-23,
    // lanes 16-31 hold K 8-15/24-31.
    union { v16bf v; v8bf h[2]; } bfrag[2];
#pragma unroll
    for (int nt = 0; nt < 2; ++nt) {
      const unsigned short* p =
          lb + (wn * 32 + nt * 16 + l16) * LDSS + halfsel * 8;
      bfrag[nt].h[0] = *(const v8bf*)(p);
      bfrag[nt].h[1] = *(const v8bf*)(p + 16);
    }

#pragma unroll
    for (int mt = 0; mt < 4; ++mt) {
#pragma unroll
      for (int nt = 0; nt < 2; ++nt) {
        acc[mt][nt] = __builtin_amdgcn_wmma_f32_16x16x32_bf16(
            false, afrag[mt].v, false, bfrag[nt].v,
            (short)0, acc[mt][nt], false, false);
      }
    }

    // Publish the next buffer: DMA must be complete before anyone reads it.
    if (issuer) __builtin_amdgcn_s_wait_tensorcnt(0);
    __syncthreads();
  }

  // Epilogue: D layout -> VGPR r holds M = r + 8*halfsel, N = l16.
#pragma unroll
  for (int nt = 0; nt < 2; ++nt) {
    const int n = n0 + wn * 32 + nt * 16 + l16;
    const float s = scale[n];
#pragma unroll
    for (int mt = 0; mt < 4; ++mt) {
      const int mbase = m0 + wm * 64 + mt * 16 + halfsel * 8;
#pragma unroll
      for (int r = 0; r < 8; ++r) {
        out[(size_t)(mbase + r) * N_ + n] = acc[mt][nt][r] * s;
      }
    }
  }
}

// ---------------------------------------------------------------------------
extern "C" void kernel_launch(void* const* d_in, const int* in_sizes, int n_in,
                              void* d_out, int out_size, void* d_ws, size_t ws_size,
                              hipStream_t stream) {
  (void)in_sizes; (void)n_in; (void)out_size; (void)ws_size;
  const float* x = (const float*)d_in[0];
  const float* w = (const float*)d_in[1];
  float* out = (float*)d_out;

  // workspace: [ x_bf16 : 32MB ][ wq_bf16 : 32MB ][ scale : 32KB ]
  unsigned short* xq = (unsigned short*)d_ws;
  unsigned short* wq = xq + (size_t)M_ * K_;
  float* scale = (float*)(wq + (size_t)N_ * K_);

  bitlinear_quant_w<<<N_, 256, 0, stream>>>(w, wq, scale);
  bitlinear_cvt_x<<<(M_ * K_) / (256 * 8), 256, 0, stream>>>(x, xq);

  dim3 grid(N_ / BT, M_ / BT);
  bitlinear_gemm<<<grid, 256, 0, stream>>>(xq, wq, scale, out);
}
